// SSDDefaultLoss_61821759259088
// MI455X (gfx1250) — compile-verified, matrix operand order checked by reading, not run on hardware
//
#include <hip/hip_runtime.h>
#include <hip/hip_bf16.h>

#define BB   64
#define AA   8732
#define CC   81
#define NGTD 100
#define RATIO 3

#define TILE_M 16
#define WAVES  8
#define TILES_PER_ROW ((AA + TILE_M - 1) / TILE_M)      /* 546 */
#define BLOCKS_X ((TILES_PER_ROW + WAVES - 1) / WAVES)  /* 69  */
#define NBLK (BLOCKS_X * BB)                            /* 4416 */

typedef _Float16 v16h __attribute__((ext_vector_type(16)));
typedef float    v8f  __attribute__((ext_vector_type(8)));
typedef unsigned int v4u __attribute__((ext_vector_type(4)));
typedef int      v8i  __attribute__((ext_vector_type(8)));
typedef int      v4i  __attribute__((ext_vector_type(4)));

__device__ __forceinline__ unsigned int orderKey(float f) {
    unsigned int u = __float_as_uint(f);
    return (u & 0x80000000u) ? ~u : (u | 0x80000000u);
}

// ---------------------------------------------------------------------------
// Kernel 0: zero the per-row foreground counters (re-run every launch).
// ---------------------------------------------------------------------------
__global__ void ssd_init(int* __restrict__ fgRow) {
    int i = threadIdx.x;
    if (i < BB) fgRow[i] = 0;
}

// ---------------------------------------------------------------------------
// Kernel 1: per-anchor cross-entropy (TDM tile load -> WMMA exp-sum) +
//           foreground smooth-L1 partials. One 16-anchor tile per wave.
// ---------------------------------------------------------------------------
__global__ void __launch_bounds__(256) ssd_main(
    const float* __restrict__ tboxes,   // [B,NGT,4]
    const int*   __restrict__ tlabels,  // [B,NGT]
    const float* __restrict__ breg,     // [B,A,4]
    const float* __restrict__ logits,   // [B,A,C]
    const float* __restrict__ anchors,  // [B,A,4]
    const int*   __restrict__ midx,     // [B,A]
    float* __restrict__ negloss,        // [B,A]  (-inf on fg)
    int*   __restrict__ fgRow,          // [B]
    float* __restrict__ blkBbox,        // [NBLK]
    float* __restrict__ blkCls)         // [NBLK]
{
    __shared__ float tile[WAVES][TILE_M * CC];   // 8 * 5184B
    __shared__ float mxs [WAVES][TILE_M];
    __shared__ float sms [WAVES][TILE_M];
    __shared__ float redF[256];
    __shared__ float redC[256];
    __shared__ int   redI[256];

    const int tid  = threadIdx.x;
    const int wave = tid >> 5;
    const int lane = tid & 31;
    const int b    = blockIdx.y;

    int tIdx = blockIdx.x * WAVES + wave;
    const bool valid = (tIdx < TILES_PER_ROW);
    if (!valid) tIdx = TILES_PER_ROW - 1;          // idle waves redo a valid tile
    const int  a0 = tIdx * TILE_M;
    const long long g0 = (long long)b * AA + a0;   // global anchor-row of tile start
    const long long totalRows = (long long)BB * AA;
    int rows = TILE_M;
    if (g0 + rows > totalRows) rows = (int)(totalRows - g0);   // last tile of last row

    // ---- TDM: DMA a [rows x 81] f32 tile (row-major, stride 81) into LDS ----
    unsigned int ldsOff = (unsigned int)(unsigned long long)(&tile[wave][0]); // LDS byte offset
    unsigned long long gaddr =
        (unsigned long long)logits + (unsigned long long)g0 * (CC * 4ull);

    v4u d0; v8i d1; v4i d2 = {0, 0, 0, 0}; v4i d3 = {0, 0, 0, 0};
    v8i dext = {0, 0, 0, 0, 0, 0, 0, 0};
    d0[0] = 1u;                                             // count=1, user descriptor
    d0[1] = ldsOff;                                         // lds_addr
    d0[2] = (unsigned int)(gaddr & 0xFFFFFFFFull);          // global_addr[31:0]
    d0[3] = (unsigned int)((gaddr >> 32) & 0x01FFFFFFull) | 0x80000000u; // addr[56:32] | type=2
    d1[0] = (int)(2u << 16);                                // data_size = 4B
    d1[1] = (int)(((unsigned)CC & 0xFFFFu) << 16);          // tensor_dim0 lo16 @ [63:48]
    d1[2] = (int)((((unsigned)totalRows) & 0xFFFFu) << 16); // dim0 hi16(=0) | dim1 lo16
    d1[3] = (int)((((unsigned)totalRows >> 16) & 0xFFFFu) | ((unsigned)CC << 16)); // dim1 hi16 | tile_dim0
    d1[4] = rows;                                           // tile_dim1 (tile_dim2=0)
    d1[5] = CC;                                             // tensor_dim0_stride lo32
    d1[6] = 0;
    d1[7] = 0;
    __builtin_amdgcn_tensor_load_to_lds(d0, d1, d2, d3, dext, 0);
    __builtin_amdgcn_s_wait_tensorcnt(0);

    float* T = &tile[wave][0];

    // ---- per-anchor max over the 81 classes (lanes 0..15, one anchor each) ----
    if (lane < TILE_M) {
        float m = -__builtin_inff();
        for (int k = 0; k < CC; ++k) m = fmaxf(m, T[lane * CC + k]);
        mxs[wave][lane] = m;
    }

    // ---- Sum(exp) via 3x v_wmma_f32_16x16x32_f16 against an all-ones B ----
    const int   m    = lane & 15;
    const int   kb   = (lane < 16) ? 0 : 8;   // 16-bit A-matrix lane->K mapping
    const float mval = mxs[wave][m];

    v16h ones;
#pragma unroll
    for (int j = 0; j < 16; ++j) ones[j] = (_Float16)1.0f;

    v8f acc = {};
#pragma unroll
    for (int c = 0; c < 3; ++c) {
        const int cb = c * 32;
        v16h afrag;
#pragma unroll
        for (int j = 0; j < 8; ++j) {
            int k1 = cb + kb + j;                                  // K = kb..kb+7
            float e1 = (k1 < CC) ? __expf(T[m * CC + k1] - mval) : 0.0f;
            afrag[j] = (_Float16)e1;
            int k2 = cb + kb + 16 + j;                             // K = kb+16..kb+23
            float e2 = (k2 < CC) ? __expf(T[m * CC + k2] - mval) : 0.0f;
            afrag[8 + j] = (_Float16)e2;
        }
        acc = __builtin_amdgcn_wmma_f32_16x16x32_f16(
                  false, afrag, false, ones, (short)0, acc, false, false);
    }
    // D layout: lane 0 holds anchors 0..7 in acc[0..7]; lane 16 holds anchors 8..15
    if (lane == 0) {
#pragma unroll
        for (int r = 0; r < 8; ++r) sms[wave][r] = acc[r];
    }
    if (lane == 16) {
#pragma unroll
        for (int r = 0; r < 8; ++r) sms[wave][8 + r] = acc[r];
    }

    // ---- per-anchor losses (lanes 0..15) ----
    float bboxP = 0.0f, clsP = 0.0f; int fgP = 0;
    if (valid && lane < TILE_M) {
        int a = a0 + lane;
        if (a < AA) {
            long long ga = (long long)b * AA + a;
            int  mi = midx[ga];
            bool fg = (mi >= 0);
            int  gi = fg ? mi : 0;
            int  lbl = fg ? tlabels[b * NGTD + gi] : 0;
            float lse   = mxs[wave][lane] + __logf(sms[wave][lane]);
            float closs = lse - T[lane * CC + lbl];
            negloss[ga] = fg ? -__builtin_inff() : closs;
            if (fg) {
                clsP = closs; fgP = 1;
                const float* an = anchors + ga * 4;
                const float* gt = tboxes + ((long long)b * NGTD + gi) * 4;
                const float* rg = breg + ga * 4;
                float aw = an[2] - an[0], ah = an[3] - an[1];
                float ax = an[0] + 0.5f * aw, ay = an[1] + 0.5f * ah;
                float gw = gt[2] - gt[0], gh = gt[3] - gt[1];
                float gx = gt[0] + 0.5f * gw, gy = gt[1] + 0.5f * gh;
                float tv[4];
                tv[0] = 10.0f * (gx - ax) / aw;
                tv[1] = 10.0f * (gy - ay) / ah;
                tv[2] = 5.0f * __logf(gw / aw);
                tv[3] = 5.0f * __logf(gh / ah);
#pragma unroll
                for (int i = 0; i < 4; ++i) {
                    float d  = rg[i] - tv[i];
                    float ad = fabsf(d);
                    bboxP += (ad < 1.0f) ? 0.5f * d * d : ad - 0.5f;
                }
            }
        }
    }

    // ---- block reduction -> per-block partials (deterministic final sum) ----
    redF[tid] = bboxP; redC[tid] = clsP; redI[tid] = fgP;
    __syncthreads();
    for (int s = 128; s > 0; s >>= 1) {
        if (tid < s) {
            redF[tid] += redF[tid + s];
            redC[tid] += redC[tid + s];
            redI[tid] += redI[tid + s];
        }
        __syncthreads();
    }
    if (tid == 0) {
        int blk = blockIdx.y * gridDim.x + blockIdx.x;
        blkBbox[blk] = redF[0];
        blkCls[blk]  = redC[0];
        if (redI[0]) atomicAdd(&fgRow[b], redI[0]);
    }
}

// ---------------------------------------------------------------------------
// Kernel 2: per-row exact top-k sum (k = 3*#fg) via 4-pass MSB radix select
// on monotone float keys. One block per batch row; row cached in LDS.
// ---------------------------------------------------------------------------
__global__ void __launch_bounds__(256) ssd_topk(
    const float* __restrict__ negloss,
    const int*   __restrict__ fgRow,
    float* __restrict__ rowTopk)
{
    __shared__ float vals[AA];
    __shared__ unsigned int hist[256];
    __shared__ unsigned int s_prefix, s_mask;
    __shared__ int s_krem;
    __shared__ float s_tval;
    __shared__ float red[256];

    const int b   = blockIdx.x;
    const int tid = threadIdx.x;

    int k = RATIO * fgRow[b];
    if (k <= 0) { if (tid == 0) rowTopk[b] = 0.0f; return; }
    if (k > AA) k = AA;

    for (int i = tid; i < AA; i += 256) vals[i] = negloss[(long long)b * AA + i];
    if (tid == 0) { s_prefix = 0u; s_mask = 0u; s_krem = k; }
    __syncthreads();

    for (int p = 3; p >= 0; --p) {
        hist[tid] = 0u;
        __syncthreads();
        unsigned int pf = s_prefix, mk = s_mask;
        for (int i = tid; i < AA; i += 256) {
            unsigned int key = orderKey(vals[i]);
            if ((key & mk) == pf)
                atomicAdd(&hist[(key >> (8 * p)) & 255u], 1u);
        }
        __syncthreads();
        if (tid == 0) {
            int kr = s_krem;
            unsigned int cum = 0; int sel = 0; unsigned int above = 0;
            for (int bin = 255; bin >= 0; --bin) {
                unsigned int h = hist[bin];
                if (cum + h >= (unsigned int)kr) { sel = bin; above = cum; break; }
                cum += h;
            }
            s_prefix = pf | ((unsigned int)sel << (8 * p));
            s_mask   = mk | (255u << (8 * p));
            s_krem   = kr - (int)above;
        }
        __syncthreads();
    }

    const unsigned int Tk = s_prefix;
    const int krem = s_krem;
    float accv = 0.0f;
    for (int i = tid; i < AA; i += 256) {
        float v = vals[i];
        unsigned int key = orderKey(v);
        if (key >  Tk) accv += v;
        if (key == Tk) s_tval = v;   // benign race: identical bit patterns
    }
    red[tid] = accv;
    __syncthreads();
    for (int s = 128; s > 0; s >>= 1) {
        if (tid < s) red[tid] += red[tid + s];
        __syncthreads();
    }
    if (tid == 0) rowTopk[b] = red[0] + (float)krem * s_tval;
}

// ---------------------------------------------------------------------------
// Kernel 3: deterministic fixed-order final reduction -> 2 scalars.
// ---------------------------------------------------------------------------
__global__ void ssd_final(const int* __restrict__ fgRow,
                          const float* __restrict__ rowTopk,
                          const float* __restrict__ blkBbox,
                          const float* __restrict__ blkCls,
                          float* __restrict__ out)
{
    if (threadIdx.x == 0 && blockIdx.x == 0) {
        int tot = 0;
        for (int i = 0; i < BB; ++i) tot += fgRow[i];
        float mined = 0.0f;
        for (int i = 0; i < BB; ++i) mined += rowTopk[i];
        float bb = 0.0f, cl = 0.0f;
        for (int i = 0; i < NBLK; ++i) { bb += blkBbox[i]; cl += blkCls[i]; }
        float N = (float)(tot > 0 ? tot : 1);
        out[0] = bb / N;
        out[1] = (cl + mined) / N;
    }
}

// ---------------------------------------------------------------------------
extern "C" void kernel_launch(void* const* d_in, const int* in_sizes, int n_in,
                              void* d_out, int out_size, void* d_ws, size_t ws_size,
                              hipStream_t stream)
{
    const float* tboxes  = (const float*)d_in[0];
    const int*   tlabels = (const int*)  d_in[1];
    const float* breg    = (const float*)d_in[2];
    const float* logits  = (const float*)d_in[3];
    const float* anchors = (const float*)d_in[4];
    const int*   midx    = (const int*)  d_in[5];
    float* out = (float*)d_out;

    float* negloss = (float*)d_ws;                       // B*A floats
    int*   fgRow   = (int*)(negloss + (size_t)BB * AA);  // B ints
    float* rowTopk = (float*)(fgRow + BB);               // B floats
    float* blkBbox = rowTopk + BB;                       // NBLK floats
    float* blkCls  = blkBbox + NBLK;                     // NBLK floats

    ssd_init<<<dim3(1), dim3(64), 0, stream>>>(fgRow);
    ssd_main<<<dim3(BLOCKS_X, BB), dim3(256), 0, stream>>>(
        tboxes, tlabels, breg, logits, anchors, midx,
        negloss, fgRow, blkBbox, blkCls);
    ssd_topk<<<dim3(BB), dim3(256), 0, stream>>>(negloss, fgRow, rowTopk);
    ssd_final<<<dim3(1), dim3(64), 0, stream>>>(fgRow, rowTopk, blkBbox, blkCls, out);
}
